// FARNN_S_D_W_45801531244717
// MI455X (gfx1250) — matrix operand order repaired and checked by reading, not compiled, not run on hardware
//
#include <hip/hip_runtime.h>

// ---------------- problem dimensions ----------------
#define VOCAB_N 30000
#define D_DIM   100
#define R_DIM   150
#define RW_DIM  50
#define S_DIM   50
#define C_DIM   20
#define B_DIM   256
#define L_DIM   512
// padded (WMMA-tile friendly)
#define DP 128
#define RP 160
#define RWP 64
#define SP 64
#define CP 32

// ---------------- workspace layout (bytes) ----------------
// consts (f32)
#define OFS_CS    0ull            // C_vec_sum  f32[RP]
#define OFS_U     640ull          // u          f32[RP]
#define OFS_UW    1280ull         // uw         f32[RWP]
#define OFS_CWS   1536ull         // Cw_sum     f32[RWP]
#define OFS_WILD  1792ull         // wild_mat   f32[64*64] padded
// bf16 B-operand fragment arrays (each fragment = 32 lanes * 16 bf16 = 1KB)
#define FR_S1     18432ull        // 2x10 frags (S1:   [S=50 x R=150])
#define FR_S1W    38912ull        // 2x4        (S1_w: [50 x 50])
#define FR_WSS    47104ull        // 2x4        (Wss1: [50 x 50])
#define FR_WRS    55296ull        // 5x4        (Wrs1: [150 x 50])
#define FR_S2T    75776ull        // 5x4        (S2^T: [150 x 50])
#define FR_WILD   96256ull        // 2x4        (wild: [50 x 50])
#define FR_CET    104448ull       // 5x2        (C_embed^T: [150 x 20])
#define FR_CWT    114688ull       // 2x2        (C_w^T:     [50 x 20])
#define FR_EMB    118784ull       // 4x10       (embed_r:   [100 x 150])
// big streaming arrays, layout [t][b][RP] f32
#define OFS_VV    163840ull
#define OFS_RA    (163840ull + 83886080ull)   // + L*B*RP*4

typedef __attribute__((ext_vector_type(16))) __bf16 v16bf;
typedef __attribute__((ext_vector_type(8)))  float  v8f;

__device__ inline v8f wmma_bf16(v16bf a, v16bf b, v8f c) {
  // D(f32,16x16) = A(bf16,16x32) * B(bf16,32x16) + C
  return __builtin_amdgcn_wmma_f32_16x16x32_bf16(false, a, false, b, (short)0, c,
                                                 false, false);
}

__device__ inline v8f zero_v8f() {
  v8f z = {0.f, 0.f, 0.f, 0.f, 0.f, 0.f, 0.f, 0.f};
  return z;
}

// Build A fragment (16x32 bf16): lane holds row M=lane&15; element e maps to
// K = (e&7) + (e>=8 ? 16 : 0) + (lane>=16 ? 8 : 0)   (ISA 7.12.2, 16-bit A).
template <typename F>
__device__ inline v16bf build_afrag(int lane, F&& getv) {
  const int M  = lane & 15;
  const int hh = (lane >> 4) ? 8 : 0;
  v16bf a;
#pragma unroll
  for (int e = 0; e < 16; ++e) {
    int K = (e & 7) + ((e & 8) ? 16 : 0) + hh;
    a[e] = (__bf16)getv(M, K);
  }
  return a;
}

// Load a prepacked B fragment (32x16 bf16) straight into VGPRs (32B/lane).
__device__ inline v16bf load_bfrag(const __bf16* base, int frag, int lane) {
  return *(const v16bf*)(base + ((size_t)frag << 9) + ((size_t)lane << 4));
}

// Store C/D fragment to row-major f32: lane = col N, elem g = row M(+8*half).
__device__ inline void store_c(float* base, int ldc, int lane, int ncol, v8f d) {
  const int n  = (lane & 15) + ncol;
  const int r0 = (lane >> 4) ? 8 : 0;
#pragma unroll
  for (int g = 0; g < 8; ++g) base[(size_t)(g + r0) * ldc + n] = d[g];
}

// Pack a f32 matrix into B-operand bf16 fragments.
// Logical B[k][n]; src value = transpose ? src[n*ld+k] : src[k*ld+n].
// Frag layout: frag = kt*NT + nt; within frag: lane,elem with
// N = nt*16 + (lane&15), K = kt*32 + (e&7) + (e>=8 ? 16 : 0) + (lane>=16 ? 8 : 0).
__device__ void pack_bfrags(const float* src, int ld, int transpose, int Kact,
                            int Nact, int KT, int NT, __bf16* dst, int tid,
                            int nth) {
  const int total = KT * NT * 512;
  for (int idx = tid; idx < total; idx += nth) {
    int frag = idx >> 9, within = idx & 511;
    int lane = within >> 4, e = within & 15;
    int kt = frag / NT, nt = frag - kt * NT;
    int n = nt * 16 + (lane & 15);
    int k = kt * 32 + (e & 7) + ((e & 8) ? 16 : 0) + ((lane >> 4) ? 8 : 0);
    float v = 0.f;
    if (k < Kact && n < Nact)
      v = transpose ? src[(size_t)n * ld + k] : src[(size_t)k * ld + n];
    dst[idx] = (__bf16)v;
  }
}

// ---------------- kernel 0: precompute + fragment packing ----------------
__global__ void __launch_bounds__(256) FARNN_prep(
    const float* C_embed, const float* S1, const float* S2, const float* S1_w,
    const float* S2_w, const float* C_w, const float* WW, const float* hT,
    const float* Wss1, const float* Wrs1, const float* embed_r, char* ws) {
  const int tid = threadIdx.x, nth = 256;
  float* Cs   = (float*)(ws + OFS_CS);
  float* U    = (float*)(ws + OFS_U);
  float* UW   = (float*)(ws + OFS_UW);
  float* CWS  = (float*)(ws + OFS_CWS);
  float* WILD = (float*)(ws + OFS_WILD);

  for (int r = tid; r < RP; r += nth) {
    float cs = 0.f, uu = 0.f;
    if (r < R_DIM) {
      for (int c = 0; c < C_DIM; ++c) cs += C_embed[c * R_DIM + r];
      for (int s = 0; s < S_DIM; ++s) uu += hT[s] * S2[s * R_DIM + r];
    }
    Cs[r] = cs; U[r] = uu;
  }
  for (int r = tid; r < RWP; r += nth) {
    float cw = 0.f, uw = 0.f;
    if (r < RW_DIM) {
      for (int c = 0; c < C_DIM; ++c) cw += C_w[c * RW_DIM + r];
      for (int s = 0; s < S_DIM; ++s) uw += hT[s] * S2_w[s * RW_DIM + r];
    }
    CWS[r] = cw; UW[r] = uw;
  }
  __syncthreads();
  // wild_mat = S1_w diag(Cw_sum) S2_w^T + WW, padded to 64x64
  for (int idx = tid; idx < 64 * 64; idx += nth) {
    int s = idx >> 6, t2 = idx & 63;
    float v = 0.f;
    if (s < S_DIM && t2 < S_DIM) {
      float acc = WW[s * S_DIM + t2];
      for (int r = 0; r < RW_DIM; ++r)
        acc += S1_w[s * RW_DIM + r] * CWS[r] * S2_w[t2 * RW_DIM + r];
      v = acc;
    }
    WILD[idx] = v;
  }
  __syncthreads();
  pack_bfrags(S1,      R_DIM,  0, S_DIM, R_DIM,  2, 10, (__bf16*)(ws + FR_S1),  tid, nth);
  pack_bfrags(S1_w,    RW_DIM, 0, S_DIM, RW_DIM, 2, 4,  (__bf16*)(ws + FR_S1W), tid, nth);
  pack_bfrags(Wss1,    S_DIM,  0, S_DIM, S_DIM,  2, 4,  (__bf16*)(ws + FR_WSS), tid, nth);
  pack_bfrags(Wrs1,    S_DIM,  0, R_DIM, S_DIM,  5, 4,  (__bf16*)(ws + FR_WRS), tid, nth);
  pack_bfrags(S2,      R_DIM,  1, R_DIM, S_DIM,  5, 4,  (__bf16*)(ws + FR_S2T), tid, nth);
  pack_bfrags(WILD,    64,     0, S_DIM, S_DIM,  2, 4,  (__bf16*)(ws + FR_WILD),tid, nth);
  pack_bfrags(C_embed, R_DIM,  1, R_DIM, C_DIM,  5, 2,  (__bf16*)(ws + FR_CET), tid, nth);
  pack_bfrags(C_w,     RW_DIM, 1, RW_DIM,C_DIM,  2, 2,  (__bf16*)(ws + FR_CWT), tid, nth);
  pack_bfrags(embed_r, R_DIM,  0, D_DIM, R_DIM,  4, 10, (__bf16*)(ws + FR_EMB), tid, nth);
}

// ---------------- kernel 1: embedding GEMM + blend ----------------
// V_vec = V_embed[tok]*beta + tanh(W_embed[tok] @ embed_r)*(1-beta)
// R_all = V_vec * C_vec_sum ; stored as [t][b][RP] f32 (pad cols zeroed).
__global__ void __launch_bounds__(256) FARNN_embed(const int* tokens,
                                                   const float* W_embed,
                                                   const float* V_embed,
                                                   const float* beta, char* ws) {
  const int lane = threadIdx.x & 31;
  const int wave = threadIdx.x >> 5;
  const int mtile = blockIdx.x * 8 + wave;  // 1024 blocks * 8 waves = 8192 tiles
  const int half = lane >> 4;

  const float*  Cs   = (const float*)(ws + OFS_CS);
  const __bf16* fEMB = (const __bf16*)(ws + FR_EMB);
  float* VV = (float*)(ws + OFS_VV);
  float* RA = (float*)(ws + OFS_RA);

  const int   tokA = tokens[mtile * 16 + (lane & 15)];
  const float* wrow = W_embed + (size_t)tokA * D_DIM;

  v8f acc[10];
#pragma unroll
  for (int nt = 0; nt < 10; ++nt) acc[nt] = zero_v8f();

#pragma unroll
  for (int kt = 0; kt < 4; ++kt) {
    v16bf a = build_afrag(lane, [&](int /*M*/, int K) {
      int k = kt * 32 + K;
      return (k < D_DIM) ? wrow[k] : 0.f;
    });
#pragma unroll
    for (int nt = 0; nt < 10; ++nt)
      acc[nt] = wmma_bf16(a, load_bfrag(fEMB, kt * 10 + nt, lane), acc[nt]);
  }

#pragma unroll
  for (int nt = 0; nt < 10; ++nt) {
    const int   r  = nt * 16 + (lane & 15);
    const float bt = (r < R_DIM) ? beta[r] : 0.f;
    const float cs = (r < R_DIM) ? Cs[r] : 0.f;
#pragma unroll
    for (int g = 0; g < 8; ++g) {
      int m = mtile * 16 + g + (half ? 8 : 0);
      int b = m >> 9;              // m / L
      int t = m & (L_DIM - 1);     // m % L
      size_t dst = ((size_t)t * B_DIM + b) * RP + r;
      float vv = 0.f, ra = 0.f;
      if (r < R_DIM) {
        int tk = tokens[m];
        float e = tanhf(acc[nt][g]);
        vv = V_embed[(size_t)tk * R_DIM + r] * bt + e * (1.f - bt);
        ra = vv * cs;
      }
      VV[dst] = vv;
      RA[dst] = ra;
    }
  }
}

// ---------------- kernel 2: the 512-step scan ----------------
// 16 blocks x wave32; block handles batch rows [btile*16, btile*16+16).
__global__ void __launch_bounds__(32) FARNN_scan(const float* h0,
                                                 const float* bs1,
                                                 const float* prio_a,
                                                 const float* prio_b,
                                                 const char* ws, float* out) {
  const int lane = threadIdx.x & 31;
  const int btile = blockIdx.x;
  const int half = lane >> 4;

  const float*  U    = (const float*)(ws + OFS_U);
  const float*  UW   = (const float*)(ws + OFS_UW);
  const __bf16* fS1  = (const __bf16*)(ws + FR_S1);
  const __bf16* fS1W = (const __bf16*)(ws + FR_S1W);
  const __bf16* fWSS = (const __bf16*)(ws + FR_WSS);
  const __bf16* fWRS = (const __bf16*)(ws + FR_WRS);
  const __bf16* fS2T = (const __bf16*)(ws + FR_S2T);
  const __bf16* fWLD = (const __bf16*)(ws + FR_WILD);
  const __bf16* fCET = (const __bf16*)(ws + FR_CET);
  const __bf16* fCWT = (const __bf16*)(ws + FR_CWT);
  const float*  VV   = (const float*)(ws + OFS_VV);
  const float*  RA   = (const float*)(ws + OFS_RA);

  __shared__ float hS[16][SP];      // current hidden state (pad cols = 0)
  __shared__ float hS1s[16][RP];    // h @ S1
  __shared__ float hS1ws[16][RWP];  // h @ S1_w
  __shared__ float ztp[16][SP];     // gate pre-activation
  __shared__ float hnw[16][SP];     // candidate hidden

  for (int idx = lane; idx < 16 * SP; idx += 32) {
    int m = idx >> 6, s = idx & 63;
    hS[m][s] = (s < S_DIM) ? h0[s] : 0.f;
  }

  for (int t = 0; t < L_DIM; ++t) {
    const float* vrow = VV + ((size_t)t * B_DIM + btile * 16) * RP;
    const float* rrow = RA + ((size_t)t * B_DIM + btile * 16) * RP;

    // prefetch next step's 2x10KB tile (global_prefetch_b8 hints)
    if (t + 1 < L_DIM) {
      const float* vn = VV + ((size_t)(t + 1) * B_DIM + btile * 16) * RP;
      const float* rn = RA + ((size_t)(t + 1) * B_DIM + btile * 16) * RP;
      const char* p = (const char*)((lane < 16) ? (vn + lane * RP)
                                                : (rn + (lane - 16) * RP));
#pragma unroll
      for (int o = 0; o < 640; o += 128) __builtin_prefetch(p + o, 0, 1);
    }

    // ---- A fragments of h (K = 0..63 over S padded) ----
    v16bf hA0 = build_afrag(lane, [&](int M, int K) { return hS[M][K]; });
    v16bf hA1 = build_afrag(lane, [&](int M, int K) { return hS[M][K + 32]; });

    // ---- hS1 = h @ S1  [16 x 160] ----
#pragma unroll
    for (int nt = 0; nt < 10; ++nt) {
      v8f acc = zero_v8f();
      acc = wmma_bf16(hA0, load_bfrag(fS1, 0 * 10 + nt, lane), acc);
      acc = wmma_bf16(hA1, load_bfrag(fS1, 1 * 10 + nt, lane), acc);
      store_c(&hS1s[0][0], RP, lane, nt * 16, acc);
    }
    // ---- hS1w = h @ S1_w  [16 x 64] ----
#pragma unroll
    for (int nt = 0; nt < 4; ++nt) {
      v8f acc = zero_v8f();
      acc = wmma_bf16(hA0, load_bfrag(fS1W, 0 * 4 + nt, lane), acc);
      acc = wmma_bf16(hA1, load_bfrag(fS1W, 1 * 4 + nt, lane), acc);
      store_c(&hS1ws[0][0], RWP, lane, nt * 16, acc);
    }

    // ---- A fragments of r_vec (K over R padded to 160) ----
    v16bf rA[5];
#pragma unroll
    for (int kt = 0; kt < 5; ++kt)
      rA[kt] = build_afrag(lane, [&](int M, int K) {
        return rrow[(size_t)M * RP + kt * 32 + K];
      });

    // ---- zt_pre = h @ Wss1 + r @ Wrs1 ----
#pragma unroll
    for (int nt = 0; nt < 4; ++nt) {
      v8f acc = zero_v8f();
      acc = wmma_bf16(hA0, load_bfrag(fWSS, 0 * 4 + nt, lane), acc);
      acc = wmma_bf16(hA1, load_bfrag(fWSS, 1 * 4 + nt, lane), acc);
#pragma unroll
      for (int kt = 0; kt < 5; ++kt)
        acc = wmma_bf16(rA[kt], load_bfrag(fWRS, kt * 4 + nt, lane), acc);
      store_c(&ztp[0][0], SP, lane, nt * 16, acc);
    }

    // ---- h_new = (hS1 * r) @ S2^T + h @ wild ----
    v16bf tA[5];
#pragma unroll
    for (int kt = 0; kt < 5; ++kt)
      tA[kt] = build_afrag(lane, [&](int M, int K) {
        int k = kt * 32 + K;
        return hS1s[M][k] * rrow[(size_t)M * RP + k];
      });
#pragma unroll
    for (int nt = 0; nt < 4; ++nt) {
      v8f acc = zero_v8f();
#pragma unroll
      for (int kt = 0; kt < 5; ++kt)
        acc = wmma_bf16(tA[kt], load_bfrag(fS2T, kt * 4 + nt, lane), acc);
      acc = wmma_bf16(hA0, load_bfrag(fWLD, 0 * 4 + nt, lane), acc);
      acc = wmma_bf16(hA1, load_bfrag(fWLD, 1 * 4 + nt, lane), acc);
      store_c(&hnw[0][0], SP, lane, nt * 16, acc);
    }

    // ---- scores: (hS1*v*u) @ Ce^T + (hS1w*uw) @ Cw^T ----
    v16bf sA[5];
#pragma unroll
    for (int kt = 0; kt < 5; ++kt)
      sA[kt] = build_afrag(lane, [&](int M, int K) {
        int k = kt * 32 + K;
        return hS1s[M][k] * vrow[(size_t)M * RP + k] * U[k];
      });
    v16bf wA[2];
#pragma unroll
    for (int kt = 0; kt < 2; ++kt)
      wA[kt] = build_afrag(lane, [&](int M, int K) {
        int k = kt * 32 + K;
        return hS1ws[M][k] * UW[k];
      });
#pragma unroll
    for (int nt = 0; nt < 2; ++nt) {
      v8f acc = zero_v8f();
#pragma unroll
      for (int kt = 0; kt < 5; ++kt)
        acc = wmma_bf16(sA[kt], load_bfrag(fCET, kt * 2 + nt, lane), acc);
      acc = wmma_bf16(wA[0], load_bfrag(fCWT, 0 * 2 + nt, lane), acc);
      acc = wmma_bf16(wA[1], load_bfrag(fCWT, 1 * 2 + nt, lane), acc);
      int c = nt * 16 + (lane & 15);
      if (c < C_DIM) {
        float pa = prio_a[c], pb = prio_b[c];
#pragma unroll
        for (int g = 0; g < 8; ++g) {
          int b = btile * 16 + g + (half ? 8 : 0);
          out[((size_t)b * L_DIM + t) * C_DIM + c] = acc[g] * pa + pb;
        }
      }
    }

    // ---- gated state update (f32, in LDS) ----
    for (int idx = lane; idx < 16 * SP; idx += 32) {
      int m = idx >> 6, s = idx & 63;
      if (s < S_DIM) {
        float pre = ztp[m][s] + bs1[s];
        float z = 1.f / (1.f + __expf(-pre));
        hS[m][s] = z * hnw[m][s] + (1.f - z) * hS[m][s];
      }
    }
  }
}

// ---------------- host-side launcher ----------------
extern "C" void kernel_launch(void* const* d_in, const int* in_sizes, int n_in,
                              void* d_out, int out_size, void* d_ws,
                              size_t ws_size, hipStream_t stream) {
  (void)in_sizes; (void)n_in; (void)out_size; (void)ws_size;
  const int*   tokens  = (const int*)d_in[0];
  const float* W_embed = (const float*)d_in[1];
  const float* embed_r = (const float*)d_in[2];
  const float* V_embed = (const float*)d_in[3];
  const float* C_embed = (const float*)d_in[4];
  const float* S1      = (const float*)d_in[5];
  const float* S2      = (const float*)d_in[6];
  const float* S1_w    = (const float*)d_in[7];
  const float* S2_w    = (const float*)d_in[8];
  const float* C_w     = (const float*)d_in[9];
  const float* WW      = (const float*)d_in[10];
  const float* h0      = (const float*)d_in[11];
  const float* hT      = (const float*)d_in[12];
  const float* beta    = (const float*)d_in[13];
  const float* Wss1    = (const float*)d_in[14];
  const float* Wrs1    = (const float*)d_in[15];
  const float* bs1     = (const float*)d_in[16];
  const float* prio_a  = (const float*)d_in[17];
  const float* prio_b  = (const float*)d_in[18];
  char*  ws  = (char*)d_ws;
  float* out = (float*)d_out;

  FARNN_prep<<<1, 256, 0, stream>>>(C_embed, S1, S2, S1_w, S2_w, C_w, WW, hT,
                                    Wss1, Wrs1, embed_r, ws);
  FARNN_embed<<<1024, 256, 0, stream>>>(tokens, W_embed, V_embed, beta, ws);
  FARNN_scan<<<16, 32, 0, stream>>>(h0, bs1, prio_a, prio_b, ws, out);
}
